// HyenaDnaFilter_71038759076219
// MI455X (gfx1250) — matrix operand order repaired and checked by reading, not compiled, not run on hardware
//
#include <hip/hip_runtime.h>
#include <hip/hip_bf16.h>
#include <stdint.h>

// ---------------- problem constants ----------------
#define L_LEN 8192      // sequence length
#define CHN   256       // HIDDEN channels
#define BATCH 8
#define FO    64        // FILTER_ORDER
#define NREV  8320      // 8192 + 128 zero pad (indices > 8191 read as 0 => causal mask)

typedef __attribute__((ext_vector_type(16))) _Float16 v16h;
typedef __attribute__((ext_vector_type(8)))  float    v8f;

union ABPack { unsigned int u[8]; v16h h; };

// funnel-select: odd ? halves(lo.hi, hi.lo) : lo   (per-lane constant 'odd' -> v_cndmask)
static __device__ __forceinline__ unsigned hsel(unsigned hi, unsigned lo, bool odd) {
    return odd ? ((lo >> 16) | (hi << 16)) : lo;
}

// ---------------------------------------------------------------------------
// Stage 1: MLP over positional embedding -> h3[t][o], f32.  One 64-thread
// block per position t; h shared across layers via LDS.  ~0.07 GFLOP total.
// ---------------------------------------------------------------------------
__global__ __launch_bounds__(64) void k_filter_mlp(
    const float* __restrict__ W1, const float* __restrict__ b1, const float* __restrict__ f0,
    const float* __restrict__ W2, const float* __restrict__ b2, const float* __restrict__ f1,
    const float* __restrict__ W3, const float* __restrict__ b3, const float* __restrict__ f2,
    float* __restrict__ h3out)
{
    __shared__ float sh[FO];
    const int t = blockIdx.x;
    const int o = threadIdx.x;

    // HyenaDnaPositionalEmbedding: z = [t_lin, cos(-f*w), sin(-f*w)], f = {1e-4, 1}
    const float tl = (float)t * (1.0f / (float)(L_LEN - 1));
    const float w  = 6.28318530717958647692f * (float)t * (1.0f / (float)L_LEN);
    const float z0 = tl;
    const float z1 =  cosf(1e-4f * w);
    const float z2 =  cosf(w);
    const float z3 = -sinf(1e-4f * w);
    const float z4 = -sinf(w);

    float a1 = b1[o] + z0 * W1[o * 5 + 0] + z1 * W1[o * 5 + 1] + z2 * W1[o * 5 + 2]
                     + z3 * W1[o * 5 + 3] + z4 * W1[o * 5 + 4];
    sh[o] = sinf(f0[o] * a1);
    __syncthreads();

    float a2 = b2[o];
    #pragma unroll 8
    for (int p = 0; p < FO; ++p) a2 += sh[p] * W2[o * FO + p];
    float h2 = sinf(f1[o] * a2);
    __syncthreads();
    sh[o] = h2;
    __syncthreads();

    float a3 = b3[o];
    #pragma unroll 8
    for (int p = 0; p < FO; ++p) a3 += sh[p] * W3[o * FO + p];
    h3out[t * FO + o] = sinf(f2[o] * a3);
}

// ---------------------------------------------------------------------------
// Stage 2: k[c,t] = (Wout[c,:] . h3[t,:]) * (exp(-t_lin*|delta_c|)+SHIFT),
// stored REVERSED in f16:  krev[c][i] = k[c][8191-i], zeros for i>8191.
// Reversal makes the Toeplitz A-operand dwords *ascending* contiguous halves.
// ---------------------------------------------------------------------------
__global__ __launch_bounds__(256) void k_build_krev(
    const float* __restrict__ h3f, const float* __restrict__ Wout,
    _Float16* __restrict__ krev)
{
    const int idx = blockIdx.x * 256 + threadIdx.x;   // idx = c*NREV + i
    const int c = idx / NREV;
    const int i = idx - c * NREV;
    const int t = (L_LEN - 1) - i;
    float v = 0.0f;
    if (t >= 0) {
        const float* __restrict__ ht = h3f + t * FO;
        const float* __restrict__ wo = Wout + c * FO;
        float s = 0.0f;
        #pragma unroll 8
        for (int o = 0; o < FO; ++o) s += wo[o] * ht[o];
        // deltas = linspace(ln(.01)/1.5, ln(.01)/0.3, 256)
        const float min_decay = -3.07011345732539450f;
        const float max_decay = -15.3505672866269725f;
        const float delta = min_decay + (max_decay - min_decay) * ((float)c * (1.0f / 255.0f));
        const float tlin  = (float)t * (1.0f / (float)(L_LEN - 1));
        const float env   = expf(-tlin * fabsf(delta)) + 0.05f;
        v = s * env;
    }
    krev[idx] = (_Float16)v;
}

// ---------------------------------------------------------------------------
// Stage 3: repack x (B,C,L) f32 -> xh[c][n][j] f16 so the WMMA B-operand is
// two aligned b128 loads per lane.  32 MB: stays resident in the 192 MB L2.
// ---------------------------------------------------------------------------
__global__ __launch_bounds__(256) void k_pack_x(
    const float* __restrict__ x, _Float16* __restrict__ xh)
{
    const int idx = blockIdx.x * 256 + threadIdx.x;   // c<<16 | n<<13 | j
    const int c = idx >> 16;
    const int n = (idx >> 13) & 7;
    const int j = idx & (L_LEN - 1);
    xh[idx] = (_Float16)x[(((size_t)((n << 8) | c)) << 13) | j];
}

// ---------------------------------------------------------------------------
// Stage 4: causal conv as block-Toeplitz WMMA, 4-way register-blocked.
// One wave owns 4 output tiles (64 rows) of one channel:
//   D_k(16x16) = A_k(16x32) B(32x16),  k = 0..3
//   A_k[m][kk] = k[c, (l0+16k-j0) + m - kk]   (Toeplitz, from reversed kernel)
//   B[kk][n]   = x[n, c, j0+kk]               (n = batch, cols 8..15 zeroed)
// B is shared by all 4 wmma; the 5 A half-groups overlap between tiles
// (tile k group1 == tile k-1 group0) -> 3 VMEM + 5 selects per wmma.
// Zero pad in krev handles causality and ragged per-tile chunk counts.
// ---------------------------------------------------------------------------
__global__ __launch_bounds__(256) void k_conv_wmma(
    const float* __restrict__ x, const float* __restrict__ Dv,
    const _Float16* __restrict__ krev, const _Float16* __restrict__ xh,
    float* __restrict__ y)
{
    const int wave = (blockIdx.x << 3) | (threadIdx.x >> 5);   // 8 waves / block
    const int lane = threadIdx.x & 31;
    const int c    = wave >> 7;          // 128 64-row tiles per channel
    const int l0   = (wave & 127) << 6;
    const int m    = lane & 15;          // A row (and D column index n)
    const int g    = lane >> 4;          // K-half select
    const int n    = m;                  // B / D column = batch
    const bool nv  = (n < BATCH);
    const unsigned bmask = nv ? 0xFFFFFFFFu : 0u;
    const bool odd = ((m & 1) == 0);     // group bases are odd iff m even

    const _Float16* __restrict__ kc = krev + (size_t)c * NREV;
    const _Float16* __restrict__ xc = xh + ((size_t)((c << 3) | (n & 7))) * L_LEN;

    v8f acc0 = {}, acc1 = {}, acc2 = {}, acc3 = {};

    // top group base (tile0/group1) at j0 = 0; group j sits at q_top - 16*j
    const int qtop0 = (L_LEN - 1) - l0 - m + (g << 3) + 16;
    const int jtmax = (l0 + 63) >> 5;    // covers causal range of tile3 (l0+48)

    for (int jt = 0; jt <= jtmax; ++jt) {
        const int j0 = jt << 5;
        const int hbt = (qtop0 + j0) & ~1;     // dword-aligned top group base

        // --- A: 5 overlapping 10-half groups (5x b128 + 5x b32, L2-resident) ---
        unsigned gp[5][4];
        #pragma unroll
        for (int j = 0; j < 5; ++j) {
            const unsigned* __restrict__ kdw = (const unsigned*)(kc + (hbt - 16 * j));
            const uint4    d4 = *(const uint4*)(kdw);   // halves base .. base+7
            const unsigned e  = kdw[4];                  // halves base+8, base+9
            gp[j][0] = hsel(d4.y, d4.x, odd);
            gp[j][1] = hsel(d4.z, d4.y, odd);
            gp[j][2] = hsel(d4.w, d4.z, odd);
            gp[j][3] = hsel(e,    d4.w, odd);
        }

        // --- B: x chunk, lane = batch column (2x aligned b128), shared 4x ---
        const uint4* __restrict__ xp = (const uint4*)(xc + j0 + (g << 3));
        const uint4 xb0 = xp[0];
        const uint4 xb1 = xp[2];
        ABPack pb;
        pb.u[0] = xb0.x & bmask;  pb.u[1] = xb0.y & bmask;
        pb.u[2] = xb0.z & bmask;  pb.u[3] = xb0.w & bmask;
        pb.u[4] = xb1.x & bmask;  pb.u[5] = xb1.y & bmask;
        pb.u[6] = xb1.z & bmask;  pb.u[7] = xb1.w & bmask;

        // --- 4 wmma: tile k uses group k+1 (kk 0..7|8g) and group k (kk 16..23|8g) ---
        ABPack pa;
        #pragma unroll
        for (int i = 0; i < 4; ++i) { pa.u[i] = gp[1][i]; pa.u[4 + i] = gp[0][i]; }
        acc0 = __builtin_amdgcn_wmma_f32_16x16x32_f16(false, pa.h, false, pb.h, (short)0, acc0, false, false);
        #pragma unroll
        for (int i = 0; i < 4; ++i) { pa.u[i] = gp[2][i]; pa.u[4 + i] = gp[1][i]; }
        acc1 = __builtin_amdgcn_wmma_f32_16x16x32_f16(false, pa.h, false, pb.h, (short)0, acc1, false, false);
        #pragma unroll
        for (int i = 0; i < 4; ++i) { pa.u[i] = gp[3][i]; pa.u[4 + i] = gp[2][i]; }
        acc2 = __builtin_amdgcn_wmma_f32_16x16x32_f16(false, pa.h, false, pb.h, (short)0, acc2, false, false);
        #pragma unroll
        for (int i = 0; i < 4; ++i) { pa.u[i] = gp[4][i]; pa.u[4 + i] = gp[3][i]; }
        acc3 = __builtin_amdgcn_wmma_f32_16x16x32_f16(false, pa.h, false, pb.h, (short)0, acc3, false, false);
    }

    // Epilogue: D-matrix layout lane n = col, VGPR r -> row r + 8g.  y += D[c]*x.
    if (nv) {
        const float dcoef = Dv[c];
        const size_t cb = (size_t)((n << 8) | c) * L_LEN + (size_t)(l0 + (g << 3));
        const float* __restrict__ xg = x + cb;
        float* __restrict__ yg = y + cb;
        #pragma unroll
        for (int r = 0; r < 8; ++r) yg[r]      = acc0[r] + dcoef * xg[r];
        #pragma unroll
        for (int r = 0; r < 8; ++r) yg[16 + r] = acc1[r] + dcoef * xg[16 + r];
        #pragma unroll
        for (int r = 0; r < 8; ++r) yg[32 + r] = acc2[r] + dcoef * xg[32 + r];
        #pragma unroll
        for (int r = 0; r < 8; ++r) yg[48 + r] = acc3[r] + dcoef * xg[48 + r];
    }
}

// ---------------------------------------------------------------------------
extern "C" void kernel_launch(void* const* d_in, const int* in_sizes, int n_in,
                              void* d_out, int out_size, void* d_ws, size_t ws_size,
                              hipStream_t stream)
{
    const float* x    = (const float*)d_in[0];
    const float* W1   = (const float*)d_in[1];
    const float* b1   = (const float*)d_in[2];
    const float* f0   = (const float*)d_in[3];
    const float* W2   = (const float*)d_in[4];
    const float* b2   = (const float*)d_in[5];
    const float* f1   = (const float*)d_in[6];
    const float* W3   = (const float*)d_in[7];
    const float* b3   = (const float*)d_in[8];
    const float* f2   = (const float*)d_in[9];
    const float* Wout = (const float*)d_in[10];
    const float* Dv   = (const float*)d_in[11];
    float* y = (float*)d_out;

    char* ws = (char*)d_ws;
    float*    h3f  = (float*)ws;                                   // 8192*64*4  = 2 MiB
    _Float16* krev = (_Float16*)(ws + (size_t)L_LEN * FO * 4);     // 256*8320*2 = 4.06 MiB
    _Float16* xh   = (_Float16*)(ws + (size_t)L_LEN * FO * 4
                                    + (size_t)CHN * NREV * 2);     // 256*8*8192*2 = 32 MiB

    k_filter_mlp<<<L_LEN, FO, 0, stream>>>(W1, b1, f0, W2, b2, f1, W3, b3, f2, h3f);
    k_build_krev<<<(CHN * NREV) / 256, 256, 0, stream>>>(h3f, Wout, krev);
    k_pack_x<<<(CHN * BATCH * L_LEN) / 256, 256, 0, stream>>>(x, xh);
    // 256 channels * 128 tiles-of-64-rows = 32768 waves, 8 waves / block
    k_conv_wmma<<<(CHN * (L_LEN / 64)) / 8, 256, 0, stream>>>(x, Dv, krev, xh, y);
}
